// SparseEncoder_29746943492352
// MI455X (gfx1250) — matrix operand order repaired and checked by reading, not compiled, not run on hardware
//
#include <hip/hip_runtime.h>

// ---------------------------------------------------------------------------
// Sparse 3D conv encoder for MI455X (gfx1250, wave32, WMMA).
// Full on-device pipeline: hash rulebook -> gather/WMMA/scatter -> BN+ReLU
// -> dense scatter. GEMMs use v_wmma_f32_16x16x32_f16 (f16 in, f32 accum).
//
// Conv kernel: 128 gathered rows per block staged once into LDS, weights
// staged once per block, both pre-swizzled into WMMA per-lane fragment
// layout so each fragment is a single 32B ds_load. Wave w owns row-tile w
// and sweeps all 16-wide column tiles, so all 8 waves do WMMA for any Cout.
// ---------------------------------------------------------------------------

#define HBITS 20
#define HSIZE (1u << HBITS)
#define HMASK (HSIZE - 1u)
#define N_MAXV 240000
#define C_MAX 128
#define KOFF_MAX 27
#define NB ((N_MAXV + 255) / 256)
#define BN_EPS 0.001f

typedef __attribute__((ext_vector_type(16))) _Float16 v16h;
typedef __attribute__((ext_vector_type(8)))  float    v8f;

__device__ __forceinline__ unsigned int hash_key(long long key) {
  unsigned long long k = (unsigned long long)key;
  k *= 0x9E3779B97F4A7C15ull;
  return (unsigned int)(k >> 40) & HMASK;
}

// ------------------------------ utility kernels ----------------------------

__global__ void k_fill_f32(float* p, float v, int n) {
  int i = blockIdx.x * blockDim.x + threadIdx.x;
  if (i < n) p[i] = v;
}
__global__ void k_fill_i32(int* p, int v, int n) {
  int i = blockIdx.x * blockDim.x + threadIdx.x;
  if (i < n) p[i] = v;
}
__global__ void k_fill_i64(long long* p, long long v, int n) {
  int i = blockIdx.x * blockDim.x + threadIdx.x;
  if (i < n) p[i] = v;
}
__global__ void k_set_int(int* p, int v) { *p = v; }
__global__ void k_copy_int(int* d, const int* s) { *d = *s; }
__global__ void k_cvt_f16(const float* __restrict__ s, _Float16* __restrict__ d, int n) {
  int i = blockIdx.x * blockDim.x + threadIdx.x;
  if (i < n) d[i] = (_Float16)s[i];
}

// ------------------------------ rulebook kernels ---------------------------

__global__ void k_hash_insert(const int* __restrict__ coords, const int* __restrict__ nin,
                              long long* __restrict__ keys, int* __restrict__ vals,
                              int d0, int d1, int d2) {
  int r = blockIdx.x * blockDim.x + threadIdx.x;
  if (r >= *nin) return;
  long long b = coords[4 * r + 0];
  long long key = ((b * d0 + coords[4 * r + 1]) * d1 + coords[4 * r + 2]) * d2 + coords[4 * r + 3];
  unsigned int h = hash_key(key);
  for (;;) {
    unsigned long long prev =
        atomicCAS((unsigned long long*)&keys[h], ~0ull, (unsigned long long)key);
    if (prev == ~0ull) { vals[h] = r; return; }
    if (prev == (unsigned long long)key) return;  // duplicate coordinate
    h = (h + 1) & HMASK;
  }
}

// Strided layers: dedupe candidate output coords, claim compacted indices.
__global__ void k_gen_out(const int* __restrict__ coords, const int* __restrict__ nin,
                          int* __restrict__ ocoords, int* __restrict__ nout,
                          long long* __restrict__ keys, int* __restrict__ vals,
                          int k0, int k1, int k2, int s0, int s1, int s2,
                          int p0, int p1, int p2, int o0, int o1, int o2) {
  int r = blockIdx.x * blockDim.x + threadIdx.x;
  if (r >= *nin) return;
  int b = coords[4 * r + 0];
  int x = coords[4 * r + 1], y = coords[4 * r + 2], z = coords[4 * r + 3];
  for (int a = 0; a < k0; ++a)
    for (int bb = 0; bb < k1; ++bb)
      for (int cc = 0; cc < k2; ++cc) {
        int nx = x + p0 - a, ny = y + p1 - bb, nz = z + p2 - cc;
        if ((nx % s0) || (ny % s1) || (nz % s2)) continue;
        int qx = nx / s0, qy = ny / s1, qz = nz / s2;
        if (qx < 0 || qx >= o0 || qy < 0 || qy >= o1 || qz < 0 || qz >= o2) continue;
        long long key = (((long long)b * o0 + qx) * o1 + qy) * o2 + qz;
        unsigned int h = hash_key(key);
        for (;;) {
          unsigned long long prev =
              atomicCAS((unsigned long long*)&keys[h], ~0ull, (unsigned long long)key);
          if (prev == ~0ull) {
            int idx = atomicAdd(nout, 1);
            vals[h] = idx;
            ocoords[4 * idx + 0] = b;
            ocoords[4 * idx + 1] = qx;
            ocoords[4 * idx + 2] = qy;
            ocoords[4 * idx + 3] = qz;
            break;
          }
          if (prev == (unsigned long long)key) break;
          h = (h + 1) & HMASK;
        }
      }
}

__global__ void k_build_pairs(const int* __restrict__ ocoords, const int* __restrict__ nout,
                              const long long* __restrict__ keys, const int* __restrict__ vals,
                              int* __restrict__ cnt, int* __restrict__ oi, int* __restrict__ ii,
                              int maxM, int i0, int i1, int i2,
                              int k0, int k1, int k2, int s0, int s1, int s2,
                              int p0, int p1, int p2) {
  int o = blockIdx.x * blockDim.x + threadIdx.x;
  if (o >= *nout) return;
  int k = blockIdx.y;
  int ka = k / (k1 * k2);
  int krem = k - ka * (k1 * k2);
  int kb = krem / k2;
  int kc = krem - kb * k2;
  int b = ocoords[4 * o + 0];
  int ix = ocoords[4 * o + 1] * s0 - p0 + ka;
  int iy = ocoords[4 * o + 2] * s1 - p1 + kb;
  int iz = ocoords[4 * o + 3] * s2 - p2 + kc;
  if (ix < 0 || ix >= i0 || iy < 0 || iy >= i1 || iz < 0 || iz >= i2) return;
  long long key = (((long long)b * i0 + ix) * i1 + iy) * i2 + iz;
  unsigned int h = hash_key(key);
  for (;;) {
    long long kk = keys[h];
    if (kk == -1ll) return;
    if (kk == key) {
      int pos = atomicAdd(&cnt[k], 1);
      oi[(size_t)k * maxM + pos] = o;
      ii[(size_t)k * maxM + pos] = vals[h];
      return;
    }
    h = (h + 1) & HMASK;
  }
}

// ------------------------------ WMMA conv kernel ---------------------------
// Block = 256 threads = 8 wave32, 128 pair-rows per block.
//
// LDS A panel: 128 rows x Cinp channels, pre-swizzled into the WMMA 16-bit
// A-matrix per-lane layout (ISA 7.12.2): element (m, k) of a 16x32 tile goes
// to lane ((k&15)>>3)*16 + m, fragment half 2*((k>>4)*4 + ((k&7)>>1)) + (k&1).
// LDS B panel: Cinp x Cout weights, pre-swizzled into the B 32x16 layout:
// element (kr, n) -> lane ((kr>>3)&1)*16 + (n&15), fragment half (kr>>4)*8 + (kr&7).
// Each fragment is then one aligned 32-byte LDS read per lane.

__global__ __launch_bounds__(256) void k_conv_wmma(
    const _Float16* __restrict__ feats, const _Float16* __restrict__ wts,
    const int* __restrict__ oi, const int* __restrict__ ii,
    const int* __restrict__ cnt, int maxM, int Cin, int Cout,
    float* __restrict__ out) {
  __shared__ _Float16 As[8192];  // 8 row-tiles x 2 K-chunks x 32 lanes x 16 halves
  __shared__ _Float16 Bs[8192];  // 2 K-chunks x 8 col-tiles x 32 lanes x 16 halves

  const int k = blockIdx.y;
  const int Mk = cnt[k];
  const int row0 = blockIdx.x * 128;
  if (row0 >= Mk) return;

  const int tid = threadIdx.x;
  const int lane = tid & 31;
  const int wave = tid >> 5;
  const int Cinp = (Cin + 31) & ~31;
  const int nchunk = Cinp >> 5;        // 1 or 2
  const int ncol = Cout >> 4;          // 1,2,4,8

  const int* oik = oi + (size_t)k * maxM;
  const int* iik = ii + (size_t)k * maxM;

  // speculative prefetch of the next row-block's pair lists
  if (tid == 0) {
    __builtin_prefetch(oik + row0 + 128, 0, 1);
    __builtin_prefetch(iik + row0 + 128, 0, 1);
  }

  // ---- stage B (weights, zero-padded in K) swizzled into fragment layout ----
  const _Float16* wk = wts + (size_t)k * Cin * Cout;
  for (int idx = tid; idx < Cinp * Cout; idx += 256) {
    int r = idx / Cout, c = idx - r * Cout;      // r = K row, c = out col
    _Float16 v = (r < Cin) ? wk[r * Cout + c] : (_Float16)0.f;
    int chunk = r >> 5, kr = r & 31;
    int dst = (((chunk * ncol + (c >> 4)) * 32) + ((kr >> 3) & 1) * 16 + (c & 15)) * 16 + (kr >> 4) * 8 + (kr & 7);
    Bs[dst] = v;
  }

  // ---- stage A (gathered rows, zero-padded) swizzled into fragment layout ----
  for (int idx = tid; idx < 128 * Cinp; idx += 256) {
    int r = (Cinp == 64) ? (idx >> 6) : (idx >> 5);
    int c = idx & (Cinp - 1);
    int p = row0 + r;
    _Float16 v = (_Float16)0.f;
    if (p < Mk && c < Cin) v = feats[(size_t)iik[p] * Cin + c];
    int rt = r >> 4, m = r & 15;
    int chunk = c >> 5, kk = c & 31;
    int ln = ((kk & 15) >> 3) * 16 + m;
    int fh = 2 * ((kk >> 4) * 4 + ((kk & 7) >> 1)) + (kk & 1);
    As[(((rt * 2 + chunk) * 32) + ln) * 16 + fh] = v;
  }
  __syncthreads();

  // ---- wave w owns row-tile w; A fragments + output rows loaded once ----
  v16h afr0 = *(const v16h*)&As[((wave * 2 + 0) * 32 + lane) * 16];
  v16h afr1 = (nchunk > 1) ? *(const v16h*)&As[((wave * 2 + 1) * 32 + lane) * 16]
                           : afr0;

  const int hi = lane >> 4;
  const int ncol4 = lane & 15;
  int orow[8];
#pragma unroll
  for (int v = 0; v < 8; ++v) {
    int p = row0 + wave * 16 + v + hi * 8;
    orow[v] = (p < Mk) ? oik[p] : -1;
  }

  for (int ct = 0; ct < ncol; ++ct) {   // sweep all 16-wide column tiles
    v8f accf = {};
    {
      v16h bf = *(const v16h*)&Bs[((0 * ncol + ct) * 32 + lane) * 16];
      accf = __builtin_amdgcn_wmma_f32_16x16x32_f16(
          false, afr0, false, bf, (short)0, accf, false, false);
    }
    if (nchunk > 1) {
      v16h bf = *(const v16h*)&Bs[((1 * ncol + ct) * 32 + lane) * 16];
      accf = __builtin_amdgcn_wmma_f32_16x16x32_f16(
          false, afr1, false, bf, (short)0, accf, false, false);
    }
    int col = ct * 16 + ncol4;
#pragma unroll
    for (int v = 0; v < 8; ++v) {
      if (orow[v] >= 0)
        atomicAdd(&out[(size_t)orow[v] * Cout + col], accf[v]);
    }
  }
}

// ------------------------------ batch-norm kernels -------------------------

__global__ __launch_bounds__(256) void k_bn_stats(
    const float* __restrict__ x, const int* __restrict__ np, int C,
    float* __restrict__ gsum, float* __restrict__ gsq) {
  __shared__ float ssum[C_MAX];
  __shared__ float ssq[C_MAX];
  int n = *np;
  for (int c = threadIdx.x; c < C; c += blockDim.x) { ssum[c] = 0.f; ssq[c] = 0.f; }
  __syncthreads();
  int r = blockIdx.x * blockDim.x + threadIdx.x;
  if (r < n) {
    for (int c = 0; c < C; ++c) {
      float v = x[(size_t)r * C + c];
      atomicAdd(&ssum[c], v);
      atomicAdd(&ssq[c], v * v);
    }
  }
  __syncthreads();
  for (int c = threadIdx.x; c < C; c += blockDim.x) {
    atomicAdd(&gsum[c], ssum[c]);
    atomicAdd(&gsq[c], ssq[c]);
  }
}

__global__ __launch_bounds__(256) void k_bn_apply(
    float* __restrict__ x, const int* __restrict__ np, int C,
    const float* __restrict__ gsum, const float* __restrict__ gsq,
    const float* __restrict__ gamma, const float* __restrict__ beta,
    _Float16* __restrict__ xf16) {
  int n = *np;
  int r = blockIdx.x * blockDim.x + threadIdx.x;
  if (r >= n) return;
  float inv_n = 1.0f / (float)n;
  for (int c = 0; c < C; ++c) {
    float m = gsum[c] * inv_n;
    float var = gsq[c] * inv_n - m * m;
    float y = (x[(size_t)r * C + c] - m) * rsqrtf(var + BN_EPS) * gamma[c] + beta[c];
    y = y > 0.f ? y : 0.f;
    x[(size_t)r * C + c] = y;
    xf16[(size_t)r * C + c] = (_Float16)y;
  }
}

// ------------------------------ final dense scatter ------------------------
// out[b, c*S2+z, s0, s1] = feats[row, c]

__global__ __launch_bounds__(256) void k_scatter(
    const float* __restrict__ feats, const int* __restrict__ coords,
    const int* __restrict__ np, float* __restrict__ out,
    int C, int S0, int S1, int S2) {
  int r = blockIdx.x * blockDim.x + threadIdx.x;
  if (r >= *np) return;
  int b = coords[4 * r + 0];
  int x = coords[4 * r + 1], y = coords[4 * r + 2], z = coords[4 * r + 3];
  for (int c = 0; c < C; ++c) {
    size_t o = ((((size_t)b * (C * S2) + (size_t)c * S2 + z) * S0 + x) * S1 + y);
    out[o] = feats[(size_t)r * C + c];
  }
}

// ------------------------------ host orchestration -------------------------

static inline char* wsalloc(char*& p, size_t bytes) {
  char* r = p;
  p += (bytes + 255) & ~(size_t)255;
  return r;
}

extern "C" void kernel_launch(void* const* d_in, const int* in_sizes, int n_in,
                              void* d_out, int out_size, void* d_ws, size_t ws_size,
                              hipStream_t stream) {
  (void)in_sizes; (void)n_in; (void)ws_size;

  static const int LCIN[12]  = {5, 16, 16, 32, 32, 32, 64, 64, 64, 64, 64, 64};
  static const int LCOUT[12] = {16, 16, 32, 32, 32, 64, 64, 64, 64, 64, 64, 128};
  static const int LK[12][3] = {{3,3,3},{3,3,3},{3,3,3},{3,3,3},{3,3,3},{3,3,3},
                                {3,3,3},{3,3,3},{3,3,3},{3,3,3},{3,3,3},{1,1,3}};
  static const int LS[12][3] = {{1,1,1},{1,1,1},{2,2,2},{1,1,1},{1,1,1},{2,2,2},
                                {1,1,1},{1,1,1},{2,2,2},{1,1,1},{1,1,1},{1,1,2}};
  static const int LP[12][3] = {{1,1,1},{1,1,1},{1,1,1},{1,1,1},{1,1,1},{1,1,1},
                                {1,1,1},{1,1,1},{0,1,1},{1,1,1},{1,1,1},{0,0,0}};

  int sh[13][3];
  sh[0][0] = 1440; sh[0][1] = 1440; sh[0][2] = 41;
  for (int i = 0; i < 12; ++i)
    for (int d = 0; d < 3; ++d)
      sh[i + 1][d] = (sh[i][d] + 2 * LP[i][d] - LK[i][d]) / LS[i][d] + 1;

  // --- workspace carve ---
  char* wp = (char*)d_ws;
  int*       nvox    = (int*)wsalloc(wp, 16 * sizeof(int));
  int*       pcnt    = (int*)wsalloc(wp, 32 * sizeof(int));
  long long* hkIn    = (long long*)wsalloc(wp, (size_t)HSIZE * 8);
  int*       hvIn    = (int*)wsalloc(wp, (size_t)HSIZE * 4);
  long long* hkOut   = (long long*)wsalloc(wp, (size_t)HSIZE * 8);
  int*       hvOut   = (int*)wsalloc(wp, (size_t)HSIZE * 4);
  int*       coordsA = (int*)wsalloc(wp, (size_t)N_MAXV * 4 * 4);
  int*       coordsB = (int*)wsalloc(wp, (size_t)N_MAXV * 4 * 4);
  int*       pair_oi = (int*)wsalloc(wp, (size_t)KOFF_MAX * N_MAXV * 4);
  int*       pair_ii = (int*)wsalloc(wp, (size_t)KOFF_MAX * N_MAXV * 4);
  float*     acc     = (float*)wsalloc(wp, (size_t)N_MAXV * C_MAX * 4);
  _Float16*  fA      = (_Float16*)wsalloc(wp, (size_t)N_MAXV * C_MAX * 2);
  _Float16*  fB      = (_Float16*)wsalloc(wp, (size_t)N_MAXV * C_MAX * 2);
  _Float16*  wh      = (_Float16*)wsalloc(wp, (size_t)KOFF_MAX * 64 * 128 * 2);
  float*     bnbuf   = (float*)wsalloc(wp, 2 * C_MAX * sizeof(float));

  const float* vfeat = (const float*)d_in[0];
  const int*   coors = (const int*)d_in[1];

  k_set_int<<<1, 1, 0, stream>>>(nvox, N_MAXV);
  k_cvt_f16<<<(N_MAXV * 5 + 255) / 256, 256, 0, stream>>>(vfeat, fA, N_MAXV * 5);

  const int* curC = coors;
  _Float16* curF = fA;
  _Float16* nxtF = fB;
  int* cbuf[2] = {coordsA, coordsB};
  int cflip = 0;

  for (int L = 0; L < 12; ++L) {
    const int cin = LCIN[L], cout = LCOUT[L];
    const int kk0 = LK[L][0], kk1 = LK[L][1], kk2 = LK[L][2];
    const int K = kk0 * kk1 * kk2;
    const int i0 = sh[L][0], i1 = sh[L][1], i2 = sh[L][2];
    const int o0 = sh[L + 1][0], o1 = sh[L + 1][1], o2 = sh[L + 1][2];
    const int s0 = LS[L][0], s1 = LS[L][1], s2 = LS[L][2];
    const int p0 = LP[L][0], p1 = LP[L][1], p2 = LP[L][2];
    const bool strided = (s0 != 1) || (s1 != 1) || (s2 != 1);

    // 1. hash input coords
    k_fill_i64<<<(HSIZE + 255) / 256, 256, 0, stream>>>(hkIn, -1ll, (int)HSIZE);
    k_hash_insert<<<NB, 256, 0, stream>>>(curC, nvox + L, hkIn, hvIn, i0, i1, i2);

    // 2. output coordinates
    const int* outC;
    if (strided) {
      k_fill_i64<<<(HSIZE + 255) / 256, 256, 0, stream>>>(hkOut, -1ll, (int)HSIZE);
      k_set_int<<<1, 1, 0, stream>>>(nvox + L + 1, 0);
      k_gen_out<<<NB, 256, 0, stream>>>(curC, nvox + L, cbuf[cflip], nvox + L + 1,
                                        hkOut, hvOut, kk0, kk1, kk2,
                                        s0, s1, s2, p0, p1, p2, o0, o1, o2);
      outC = cbuf[cflip];
      cflip ^= 1;
    } else {
      k_copy_int<<<1, 1, 0, stream>>>(nvox + L + 1, nvox + L);
      outC = curC;
    }

    // 3. pair lists per kernel offset
    k_fill_i32<<<1, 32, 0, stream>>>(pcnt, 0, K);
    k_build_pairs<<<dim3(NB, K), 256, 0, stream>>>(
        outC, nvox + L + 1, hkIn, hvIn, pcnt, pair_oi, pair_ii, N_MAXV,
        i0, i1, i2, kk0, kk1, kk2, s0, s1, s2, p0, p1, p2);

    // 4. weights -> f16
    const float* wL = (const float*)d_in[3 + 3 * L];
    const float* gL = (const float*)d_in[4 + 3 * L];
    const float* bL = (const float*)d_in[5 + 3 * L];
    const int wn = K * cin * cout;
    k_cvt_f16<<<(wn + 255) / 256, 256, 0, stream>>>(wL, wh, wn);

    // 5. zero accumulator, then gather-WMMA-scatter over all offsets
    const int accn = N_MAXV * cout;
    k_fill_f32<<<(accn + 255) / 256, 256, 0, stream>>>(acc, 0.f, accn);
    k_conv_wmma<<<dim3((N_MAXV + 127) / 128, K), 256, 0, stream>>>(
        curF, wh, pair_oi, pair_ii, pcnt, N_MAXV, cin, cout, acc);

    // 6. batch-norm + ReLU (writes f32 in place + f16 for next gather)
    k_fill_f32<<<1, 256, 0, stream>>>(bnbuf, 0.f, 2 * C_MAX);
    k_bn_stats<<<NB, 256, 0, stream>>>(acc, nvox + L + 1, cout, bnbuf, bnbuf + C_MAX);
    k_bn_apply<<<NB, 256, 0, stream>>>(acc, nvox + L + 1, cout, bnbuf, bnbuf + C_MAX,
                                       gL, bL, nxtF);

    _Float16* t = curF; curF = nxtF; nxtF = t;
    curC = outC;
  }

  // final dense scatter: (B, C*S2, S0, S1)
  k_fill_f32<<<(out_size + 255) / 256, 256, 0, stream>>>((float*)d_out, 0.f, out_size);
  k_scatter<<<NB, 256, 0, stream>>>(acc, curC, nvox + 12, (float*)d_out,
                                    C_MAX, sh[12][0], sh[12][1], sh[12][2]);
}
